// MultiVectorQuantizer_68925635166669
// MI455X (gfx1250) — matrix operand order repaired and checked by reading, not compiled, not run on hardware
//
#include <hip/hip_runtime.h>

typedef __bf16 bf16;
typedef __attribute__((ext_vector_type(16))) __bf16 v16bf;
typedef __attribute__((ext_vector_type(8)))  float  v8f;
typedef __attribute__((ext_vector_type(4)))  float  v4f;

#define K_CODES 512
#define D_GRP   128
#define BETA    0.25f
// tokens total = 16*2048 = 32768

// ---------------------------------------------------------------------------
// Pack codebooks into WMMA B-fragment layout (bf16 hi + bf16 lo residual).
// Fragment = one (codeTile ct of 16 codes) x (K-chunk dc of 32 dims):
//   B is 32x16 (Kwmma x N). Per-lane layout: n = lane%16 (code within tile),
//   k = 16*(lane/16) + v  (v = 0..15 packed bf16 slots).
// Storage: frag[(g*32+ct)*4+dc][lane][v] contiguous -> each lane loads its 16
// bf16 (32B) with two global_load_b128.
// ---------------------------------------------------------------------------
__global__ void vq_pack(const float* __restrict__ w1, const float* __restrict__ w2,
                        const float* __restrict__ w3, const float* __restrict__ w4,
                        bf16* __restrict__ whi, bf16* __restrict__ wlo) {
  int blk = blockIdx.x;          // 0..127 = g*32 + ct
  int g  = blk >> 5;
  int ct = blk & 31;
  const float* W = (g == 0) ? w1 : (g == 1) ? w2 : (g == 2) ? w3 : w4;
  for (int s = threadIdx.x; s < 2048; s += blockDim.x) {
    int dc  = s >> 9;            // 0..3
    int rem = s & 511;
    int l   = rem >> 4;          // lane 0..31
    int v   = rem & 15;          // slot 0..15
    int n    = l & 15;
    int code = ct * 16 + n;
    int k    = ((l >> 4) << 4) + v;     // contiguous K-halves per lane half
    int d    = dc * 32 + k;
    float val = W[code * D_GRP + d];
    bf16 h = (bf16)val;
    bf16 lo = (bf16)(val - (float)h);
    int fi = (blk * 4 + dc) * 512 + l * 16 + v;
    whi[fi] = h;
    wlo[fi] = lo;
  }
}

// ||w||^2 per code, exact f32.
__global__ void vq_wsq(const float* __restrict__ w1, const float* __restrict__ w2,
                       const float* __restrict__ w3, const float* __restrict__ w4,
                       float* __restrict__ wsq) {
  int t = blockIdx.x * blockDim.x + threadIdx.x;   // 0..2047
  if (t >= 4 * K_CODES) return;
  int g = t >> 9, code = t & 511;
  const float* W = (g == 0) ? w1 : (g == 1) ? w2 : (g == 2) ? w3 : w4;
  float s = 0.f;
  for (int d = 0; d < D_GRP; ++d) { float x = W[code * D_GRP + d]; s += x * x; }
  wsq[t] = s;
}

__global__ void vq_zero(float* p) { if (threadIdx.x == 0) *p = 0.f; }

// ---------------------------------------------------------------------------
// Main: block = 256 threads = 8 waves; block covers 32 tokens (2 tiles of 16).
// Wave w -> token tile (w>>2), group (w&3).
// Per wave: 32 code tiles * 4 K-chunks * 3 bf16 WMMAs (hi*hi + hi*lo + lo*hi).
// ---------------------------------------------------------------------------
__global__ __launch_bounds__(256) void vq_main(
    const float* __restrict__ latents,
    const float* __restrict__ w1, const float* __restrict__ w2,
    const float* __restrict__ w3, const float* __restrict__ w4,
    const bf16* __restrict__ whi, const bf16* __restrict__ wlo,
    const float* __restrict__ wsq,
    float* __restrict__ out, float* __restrict__ loss) {
  __shared__ float xs[32][512];                 // 64 KB: 32 tokens x 512 dims

  long blockTok = (long)blockIdx.x * 32;
  // Cooperative b128 copy of 32 contiguous token rows into LDS.
  {
    const v4f* src = (const v4f*)(latents + blockTok * 512);
    v4f* dst = (v4f*)&xs[0][0];
    for (int i = threadIdx.x; i < 32 * 512 / 4; i += blockDim.x) dst[i] = src[i];
  }
  __syncthreads();

  int wave = threadIdx.x >> 5;
  int lane = threadIdx.x & 31;
  int wt   = wave >> 2;                         // token tile within block
  int g    = wave & 3;                          // codebook group
  int m     = lane & 15;                        // token row in tile (A layout)
  int base8 = (lane >> 4) << 3;                 // 0 or 8; doubled to A K-half base 0/16

  // Build A fragments (contiguous K-halves, matching B packing): slot v -> k = 2*base8+v
  v16bf ah[4], al[4];
  const float* xrow = &xs[wt * 16 + m][g * D_GRP];
  #pragma unroll
  for (int dc = 0; dc < 4; ++dc) {
    #pragma unroll
    for (int v = 0; v < 16; ++v) {
      int k = (base8 << 1) + v;
      float xv = xrow[dc * 32 + k];
      bf16 h = (bf16)xv;
      ah[dc][v] = h;
      al[dc][v] = (bf16)(xv - (float)h);
    }
  }

  float bs[8]; int bi[8];
  #pragma unroll
  for (int r = 0; r < 8; ++r) { bs[r] = 3.0e38f; bi[r] = 0; }

  const bf16*  whiG = whi + (long)g * 32 * 4 * 512;
  const bf16*  wloG = wlo + (long)g * 32 * 4 * 512;
  const float* wsqG = wsq + g * K_CODES;

  for (int ct = 0; ct < 32; ++ct) {
    const bf16* ph = whiG + ct * 2048 + lane * 16;
    const bf16* pl = wloG + ct * 2048 + lane * 16;
    if (ct < 31) {                              // prefetch next code tile
      __builtin_prefetch(ph + 2048, 0, 3);
      __builtin_prefetch(pl + 2048, 0, 3);
    }
    v8f c = {};
    #pragma unroll
    for (int dc = 0; dc < 4; ++dc) {
      v16bf bh = *(const v16bf*)(ph + dc * 512);
      v16bf bl = *(const v16bf*)(pl + dc * 512);
      c = __builtin_amdgcn_wmma_f32_16x16x32_bf16(false, ah[dc], false, bh, (short)0, c, false, false);
      c = __builtin_amdgcn_wmma_f32_16x16x32_bf16(false, ah[dc], false, bl, (short)0, c, false, false);
      c = __builtin_amdgcn_wmma_f32_16x16x32_bf16(false, al[dc], false, bh, (short)0, c, false, false);
    }
    // C layout: lane -> n = lane%16, VGPR r -> token m = r + 8*(lane/16)
    int code = ct * 16 + (lane & 15);
    float sw = wsqG[code];
    #pragma unroll
    for (int r = 0; r < 8; ++r) {
      float sc = sw - 2.0f * c[r];              // dist - ||x||^2 (const per token)
      if (sc < bs[r]) { bs[r] = sc; bi[r] = code; }   // strict < == first-min
    }
  }

  // Argmin across the 16 lanes holding the same token (xor masks stay in half).
  #pragma unroll
  for (int r = 0; r < 8; ++r) {
    float s = bs[r]; int i = bi[r];
    #pragma unroll
    for (int off = 1; off < 16; off <<= 1) {
      float os = __shfl_xor(s, off, 32);
      int   oi = __shfl_xor(i, off, 32);
      if (os < s || (os == s && oi < i)) { s = os; i = oi; }
    }
    bs[r] = s; bi[r] = i;
  }

  // Gather f32 codebook rows, write q (== q_st forward), accumulate SSE.
  const float* WG = (g == 0) ? w1 : (g == 1) ? w2 : (g == 2) ? w3 : w4;
  long tokBase = blockTok + wt * 16;
  int dd = lane * 4;                            // 4 dims per lane -> b128 I/O
  float acc = 0.f;
  for (int t = 0; t < 16; ++t) {
    // tokens 0..7 live in lanes 0..15 (r=t), tokens 8..15 in lanes 16..31 (r=t-8)
    int idx = __shfl(bi[t & 7], (t >> 3) << 4, 32);
    v4f q = *(const v4f*)(WG + idx * D_GRP + dd);
    const float* xr = &xs[wt * 16 + t][g * D_GRP + dd];
    float d0 = q[0] - xr[0], d1 = q[1] - xr[1], d2 = q[2] - xr[2], d3 = q[3] - xr[3];
    acc += d0 * d0 + d1 * d1 + d2 * d2 + d3 * d3;
    *(v4f*)(out + (tokBase + t) * 512 + g * D_GRP + dd) = q;
  }
  #pragma unroll
  for (int off = 1; off < 32; off <<= 1) acc += __shfl_xor(acc, off, 32);
  if (lane == 0)
    atomicAdd(loss, acc * ((1.0f + BETA) / (16.0f * 2048.0f * 128.0f)));
}

// ---------------------------------------------------------------------------
extern "C" void kernel_launch(void* const* d_in, const int* in_sizes, int n_in,
                              void* d_out, int out_size, void* d_ws, size_t ws_size,
                              hipStream_t stream) {
  const float* latents = (const float*)d_in[0];
  const float* w1 = (const float*)d_in[1];
  const float* w2 = (const float*)d_in[2];
  const float* w3 = (const float*)d_in[3];
  const float* w4 = (const float*)d_in[4];
  float* out  = (float*)d_out;
  float* loss = out + (long)16 * 2048 * 512;    // scalar appended after q_st

  // Workspace layout: whi 512KB | wlo 512KB | wsq 8KB  (~1.03 MB total)
  bf16*  whi = (bf16*)d_ws;
  bf16*  wlo = (bf16*)((char*)d_ws + 524288);
  float* wsq = (float*)((char*)d_ws + 1048576);

  vq_zero<<<1, 32, 0, stream>>>(loss);
  vq_pack<<<128, 256, 0, stream>>>(w1, w2, w3, w4, whi, wlo);
  vq_wsq <<<8, 256, 0, stream>>>(w1, w2, w3, w4, wsq);
  vq_main<<<1024, 256, 0, stream>>>(latents, w1, w2, w3, w4, whi, wlo, wsq, out, loss);
}
